// SelfAttention_19404662243633
// MI455X (gfx1250) — compile-verified, matrix operand order checked by reading, not run on hardware
//
#include <hip/hip_runtime.h>
#include <hip/hip_bf16.h>

#define B_  2
#define T_  2048
#define C_  1024
#define H_  16
#define HD_ 64

typedef __attribute__((ext_vector_type(16))) __bf16 v16bf;
typedef __attribute__((ext_vector_type(8)))  float  v8f;
typedef __attribute__((ext_vector_type(4)))  unsigned int u32x4;

__device__ __forceinline__ __bf16 to_bf16(float f) {
    union { float f; unsigned u; } a; a.f = f;
    unsigned r = a.u + 0x7fffu + ((a.u >> 16) & 1u);   // round-to-nearest-even
    union { unsigned short s; __bf16 b; } o; o.s = (unsigned short)(r >> 16);
    return o.b;
}

// Async global -> LDS copy of 16 bytes (gfx1250 GLOBAL_LOAD_ASYNC_TO_LDS_B128,
// GV mode: 64-bit vaddr, saddr = off). Tracked by ASYNCcnt (in-order completion).
__device__ __forceinline__ void async_b128(void* lds_dst, const void* gsrc) {
    unsigned lds_off = (unsigned)(unsigned long long)lds_dst; // low 32b = LDS addr
    asm volatile("global_load_async_to_lds_b128 %0, %1, off"
                 :: "v"(lds_off), "v"(gsrc) : "memory");
}
__device__ __forceinline__ void wait_async0() {
    asm volatile("s_wait_asynccnt 0" ::: "memory");
}
__device__ __forceinline__ void wait_async1() {
    asm volatile("s_wait_asynccnt 1" ::: "memory");
}
__device__ __forceinline__ void wait_async3() {
    asm volatile("s_wait_asynccnt 3" ::: "memory");
}

// Load one 16x32 (M/N x K) bf16 fragment in the CDNA5 A/B per-lane layout.
// `row32` points at the start of this lane's 32-element K window (16B aligned).
// Lanes 0-15: k = 0..7 then 16..23 ; lanes 16-31: k = 8..15 then 24..31.
__device__ __forceinline__ v16bf load_frag(const __bf16* row32, int khalf) {
    const __bf16* p = row32 + khalf * 8;
    v16bf f;
    ((u32x4*)&f)[0] = *(const u32x4*)(p);
    ((u32x4*)&f)[1] = *(const u32x4*)(p + 16);
    return f;
}

__device__ __forceinline__ v8f wmma_bf16(v16bf a, v16bf b, v8f c) {
    return __builtin_amdgcn_wmma_f32_16x16x32_bf16(
        /*neg_a=*/false, a, /*neg_b=*/false, b,
        /*c_mod=*/(short)0, c, /*reuse_a=*/false, /*reuse_b=*/false);
}

// ---------------------------------------------------------------------------
// Kernel 0: one-shot f32 -> bf16 conversion (x and the four weight matrices).
// ---------------------------------------------------------------------------
__global__ __launch_bounds__(256) void cvt_bf16_kernel(
    const float* __restrict__ in, __bf16* __restrict__ out, int n) {
    int i = (blockIdx.x * 256 + threadIdx.x) * 8;
    if (i + 8 <= n) {
#pragma unroll
        for (int k = 0; k < 8; ++k) out[i + k] = to_bf16(in[i + k]);
    }
}

// ---------------------------------------------------------------------------
// Kernel 1: out[b,h,t,d] (bf16) = xb[b,t,:] @ Wb[n,:] + bias[n],  n = h*HD+d
// Block tile: 128 rows x 64 cols, 8 waves (wave w -> rows w*16..w*16+15).
// Double-buffered async global->LDS staging: tile j+1 DMA overlaps tile j WMMA.
// ---------------------------------------------------------------------------
__global__ __launch_bounds__(256) void proj_qkv_kernel(
    const __bf16* __restrict__ xb, const __bf16* __restrict__ Wb,
    const float* __restrict__ bias, __bf16* __restrict__ out) {
    __shared__ __bf16 As[2][128][40];   // 32 k + pad; 80B row stride (16B aligned)
    __shared__ __bf16 Bs[2][64][40];

    const int tid   = threadIdx.x;
    const int wave  = tid >> 5;
    const int lane  = tid & 31;
    const int lm    = lane & 15;
    const int khalf = lane >> 4;
    const int mbase = blockIdx.y * 128;
    const int nbase = blockIdx.x * 64;

    // staging assignments (fixed per thread)
    const int ar = tid >> 1, akk = (tid & 1) * 16;     // A: 2 thr/row, 16 elems
    const int br = tid >> 2, bkk = (tid & 3) * 8;      // B: 4 thr/row, 8 elems
    const __bf16* aSrc = xb + (size_t)(mbase + ar) * C_ + akk;
    const __bf16* bSrc = Wb + (size_t)(nbase + br) * C_ + bkk;

    v8f acc[4];
#pragma unroll
    for (int a = 0; a < 4; ++a)
#pragma unroll
        for (int i = 0; i < 8; ++i) acc[a][i] = 0.0f;

    // prologue: prefetch tile 0 into buffer 0
    async_b128(&As[0][ar][akk],     aSrc);
    async_b128(&As[0][ar][akk + 8], aSrc + 8);
    async_b128(&Bs[0][br][bkk],     bSrc);

    const int nIter = C_ / 32;
    for (int i = 0; i < nIter; ++i) {
        const int cur = i & 1;
        __syncthreads();                       // everyone done reading buf[cur^1]
        if (i + 1 < nIter) {                   // prefetch next tile into buf[cur^1]
            const int nxt = cur ^ 1;
            const int k0  = (i + 1) * 32;
            async_b128(&As[nxt][ar][akk],     aSrc + k0);
            async_b128(&As[nxt][ar][akk + 8], aSrc + k0 + 8);
            async_b128(&Bs[nxt][br][bkk],     bSrc + k0);
            wait_async3();                     // only newest 3 may stay in flight
        } else {
            wait_async0();
        }
        __syncthreads();                       // buf[cur] landed for all waves

        v16bf af = load_frag(&As[cur][wave * 16 + lm][0], khalf);
#pragma unroll
        for (int n = 0; n < 4; ++n) {
            v16bf bf = load_frag(&Bs[cur][n * 16 + lm][0], khalf);
            acc[n] = wmma_bf16(af, bf, acc[n]);
        }
    }

    // epilogue: scatter into [B,H,T,HD] bf16
#pragma unroll
    for (int a = 0; a < 4; ++a) {
        int n = nbase + a * 16 + lm;
        float bvl = bias[n];
        int h = n >> 6, d = n & 63;
#pragma unroll
        for (int r = 0; r < 8; ++r) {
            int row = mbase + wave * 16 + r + khalf * 8;   // C-layout row
            int bb = row / T_, t = row % T_;
            out[(((size_t)(bb * H_ + h)) * T_ + t) * HD_ + d] =
                to_bf16(acc[a][r] + bvl);
        }
    }
}

// ---------------------------------------------------------------------------
// Kernel 2: flash attention.  Block = (b, h, 128 q-rows); 8 waves x 16 rows.
// K tile double-buffered via async DMA; V transposed manually (fenced by the
// two existing barriers); softmax is online (no [T,T] materialization).
// ---------------------------------------------------------------------------
__global__ __launch_bounds__(256) void attn_kernel(
    const __bf16* __restrict__ Q, const __bf16* __restrict__ K,
    const __bf16* __restrict__ V, const float* __restrict__ rel,
    const int* __restrict__ mask, __bf16* __restrict__ Y) {
    __shared__ __bf16 Kt[2][32][72];     // [key][d], padded (144B stride)
    __shared__ __bf16 Vt[64][40];        // [d][key] (transposed), padded
    __shared__ __bf16 Pt[8][16][40];     // per-wave P scratch

    const int tid   = threadIdx.x;
    const int wave  = tid >> 5;
    const int lane  = tid & 31;
    const int lm    = lane & 15;
    const int khalf = lane >> 4;
    const int h     = blockIdx.y;
    const int bb    = blockIdx.z;
    const int qbase = blockIdx.x * 128 + wave * 16;
    const size_t headoff = ((size_t)(bb * H_ + h)) * T_ * HD_;
    const float NEG_INF = -__builtin_inff();

    // Q fragments stay in registers the whole time (A-layout straight from HBM)
    const __bf16* qrow = Q + headoff + (size_t)(qbase + lm) * HD_;
    v16bf qf0 = load_frag(qrow,       khalf);   // k = 0..31
    v16bf qf1 = load_frag(qrow + 32,  khalf);   // k = 32..63

    // staging assignments
    const int skey = tid >> 3, sdd = (tid & 7) * 8;    // K: 1 b128 per thread
    const __bf16* kSrc = K + headoff + (size_t)skey * HD_ + sdd;
    const int vkey = tid & 31, vd0 = (tid >> 5) * 8;   // V: 8 elems per thread
    const __bf16* vSrc = V + headoff + (size_t)vkey * HD_ + vd0;

    v8f o[4];
#pragma unroll
    for (int a = 0; a < 4; ++a)
#pragma unroll
        for (int i = 0; i < 8; ++i) o[a][i] = 0.0f;
    float mrow[8], lrow[8];
#pragma unroll
    for (int r = 0; r < 8; ++r) { mrow[r] = NEG_INF; lrow[r] = 0.0f; }

    const float scale = 0.125f;   // 1/sqrt(64)

    // prologue: prefetch K tile 0 into buffer 0
    async_b128(&Kt[0][skey][sdd], kSrc);

    const int nIter = T_ / 32;
    for (int i = 0; i < nIter; ++i) {
        const int cur = i & 1;
        const int j = i * 32;
        __syncthreads();                       // prior tile fully consumed
        if (i + 1 < nIter)                     // K DMA for next tile in flight
            async_b128(&Kt[cur ^ 1][skey][sdd],
                       kSrc + (size_t)(i + 1) * 32 * HD_);
        {   // stage V transposed for the current tile: Vt[d][key]
            const __bf16* src = vSrc + (size_t)j * HD_;
            u32x4 raw = *(const u32x4*)src;
            const __bf16* e = (const __bf16*)&raw;
#pragma unroll
            for (int k = 0; k < 8; ++k) Vt[vd0 + k][vkey] = e[k];
        }
        if (i + 1 < nIter) wait_async1(); else wait_async0();
        __syncthreads();                       // Kt[cur] + Vt ready

        // S = Q K^T : 16 x 32 tile per wave
        v8f s[2];
#pragma unroll
        for (int g = 0; g < 2; ++g) {
#pragma unroll
            for (int q = 0; q < 8; ++q) s[g][q] = 0.0f;
            v16bf kf0 = load_frag(&Kt[cur][g * 16 + lm][0],  khalf);
            v16bf kf1 = load_frag(&Kt[cur][g * 16 + lm][32], khalf);
            s[g] = wmma_bf16(qf0, kf0, s[g]);
            s[g] = wmma_bf16(qf1, kf1, s[g]);
        }

        // scale + rel_pos + mask  (element (g,r): row q, col key)
        float sv[2][8];
#pragma unroll
        for (int g = 0; g < 2; ++g)
#pragma unroll
            for (int r = 0; r < 8; ++r) {
                int q   = qbase + r + khalf * 8;
                int key = j + g * 16 + lm;
                float val = s[g][r] * scale + rel[((size_t)h * T_ + q) * T_ + key];
                if (mask[(size_t)q * T_ + key] == 1) val = NEG_INF;
                sv[g][r] = val;
            }

        // online softmax (rows live across 16-lane halves)
        float pcur[2][8];
#pragma unroll
        for (int r = 0; r < 8; ++r) {
            float mx = fmaxf(sv[0][r], sv[1][r]);
#pragma unroll
            for (int off = 8; off > 0; off >>= 1)
                mx = fmaxf(mx, __shfl_xor(mx, off, 32));
            float mnew = fmaxf(mrow[r], mx);
            float corr = __expf(mrow[r] - mnew);
            float p0 = __expf(sv[0][r] - mnew);
            float p1 = __expf(sv[1][r] - mnew);
            pcur[0][r] = p0; pcur[1][r] = p1;
            float rs = p0 + p1;
#pragma unroll
            for (int off = 8; off > 0; off >>= 1)
                rs += __shfl_xor(rs, off, 32);
            lrow[r] = lrow[r] * corr + rs;
            mrow[r] = mnew;
#pragma unroll
            for (int a = 0; a < 4; ++a) o[a][r] *= corr;
        }

        // C-layout P -> LDS -> A-layout fragment (wave-private region)
#pragma unroll
        for (int g = 0; g < 2; ++g)
#pragma unroll
            for (int r = 0; r < 8; ++r)
                Pt[wave][r + khalf * 8][g * 16 + lm] = to_bf16(pcur[g][r]);
        asm volatile("s_wait_dscnt 0" ::: "memory");
        v16bf pf = load_frag(&Pt[wave][lm][0], khalf);

        // O += P V  (B-fragments from transposed V: contiguous b128 reads)
#pragma unroll
        for (int a = 0; a < 4; ++a) {
            v16bf vf = load_frag(&Vt[a * 16 + lm][0], khalf);
            o[a] = wmma_bf16(pf, vf, o[a]);
        }
    }

    // normalize and store y[b,h,q,d] (bf16)
#pragma unroll
    for (int r = 0; r < 8; ++r) {
        float inv = 1.0f / lrow[r];
        int q = qbase + r + khalf * 8;
        __bf16* dst = Y + headoff + (size_t)q * HD_;
#pragma unroll
        for (int a = 0; a < 4; ++a)
            dst[a * 16 + lm] = to_bf16(o[a][r] * inv);
    }
}

// ---------------------------------------------------------------------------
// Kernel 3: out[b,t,n] (f32) = y'[b,t,:] @ Wp[n,:] + bp[n],
// where y'[b,t,h*HD+d] = Y[b,h,t,d] (gathered during async A staging).
// Double-buffered async staging like proj_qkv.
// ---------------------------------------------------------------------------
__global__ __launch_bounds__(256) void proj_out_kernel(
    const __bf16* __restrict__ Y, const __bf16* __restrict__ Wpb,
    const float* __restrict__ bias, float* __restrict__ out) {
    __shared__ __bf16 As[2][128][40];
    __shared__ __bf16 Bs[2][64][40];

    const int tid   = threadIdx.x;
    const int wave  = tid >> 5;
    const int lane  = tid & 31;
    const int lm    = lane & 15;
    const int khalf = lane >> 4;
    const int mbase = blockIdx.y * 128;
    const int nbase = blockIdx.x * 64;

    const int ar = tid >> 1, akk = (tid & 1) * 16;
    const int br = tid >> 2, bkk = (tid & 3) * 8;
    const int arow = mbase + ar;
    const int abb = arow / T_, at = arow % T_;
    const __bf16* yRow = Y + ((size_t)abb * H_ * T_ + at) * HD_; // + h*T*HD later
    const __bf16* bSrc = Wpb + (size_t)(nbase + br) * C_ + bkk;

    v8f acc[4];
#pragma unroll
    for (int a = 0; a < 4; ++a)
#pragma unroll
        for (int i = 0; i < 8; ++i) acc[a][i] = 0.0f;

    // A-source address for a given k0 (32-chunk stays inside one head)
    auto aSrc = [&](int k0) -> const __bf16* {
        int kg = k0 + akk;
        int hh = kg >> 6, d = kg & 63;
        return yRow + (size_t)hh * T_ * HD_ + d;
    };

    // prologue: prefetch tile 0
    async_b128(&As[0][ar][akk],     aSrc(0));
    async_b128(&As[0][ar][akk + 8], aSrc(0) + 8);
    async_b128(&Bs[0][br][bkk],     bSrc);

    const int nIter = C_ / 32;
    for (int i = 0; i < nIter; ++i) {
        const int cur = i & 1;
        __syncthreads();
        if (i + 1 < nIter) {
            const int nxt = cur ^ 1;
            const int k0  = (i + 1) * 32;
            async_b128(&As[nxt][ar][akk],     aSrc(k0));
            async_b128(&As[nxt][ar][akk + 8], aSrc(k0) + 8);
            async_b128(&Bs[nxt][br][bkk],     bSrc + k0);
            wait_async3();
        } else {
            wait_async0();
        }
        __syncthreads();

        v16bf af = load_frag(&As[cur][wave * 16 + lm][0], khalf);
#pragma unroll
        for (int n = 0; n < 4; ++n) {
            v16bf bf = load_frag(&Bs[cur][n * 16 + lm][0], khalf);
            acc[n] = wmma_bf16(af, bf, acc[n]);
        }
    }

#pragma unroll
    for (int a = 0; a < 4; ++a) {
        int n = nbase + a * 16 + lm;
        float bvl = bias[n];
#pragma unroll
        for (int r = 0; r < 8; ++r) {
            int row = mbase + wave * 16 + r + khalf * 8;
            out[(size_t)row * C_ + n] = acc[a][r] + bvl;
        }
    }
}

// ---------------------------------------------------------------------------
extern "C" void kernel_launch(void* const* d_in, const int* in_sizes, int n_in,
                              void* d_out, int out_size, void* d_ws, size_t ws_size,
                              hipStream_t stream) {
    (void)in_sizes; (void)n_in; (void)out_size; (void)ws_size;
    const float* x    = (const float*)d_in[0];
    const int*   mask = (const int*)  d_in[1];
    const float* rel  = (const float*)d_in[2];
    const float* Wq   = (const float*)d_in[3];
    const float* bq   = (const float*)d_in[4];
    const float* Wk   = (const float*)d_in[5];
    const float* bk   = (const float*)d_in[6];
    const float* Wv   = (const float*)d_in[7];
    const float* bv   = (const float*)d_in[8];
    const float* Wp   = (const float*)d_in[9];
    const float* bp   = (const float*)d_in[10];
    float* out = (float*)d_out;

    const size_t act = (size_t)B_ * H_ * T_ * HD_;     // 4,194,304 (= B*T*C)
    const size_t wel = (size_t)C_ * C_;                // 1,048,576
    __bf16* Qb  = (__bf16*)d_ws;
    __bf16* Kb  = Qb  + act;
    __bf16* Vb  = Kb  + act;
    __bf16* Yb  = Vb  + act;
    __bf16* xb  = Yb  + act;
    __bf16* Wqb = xb  + act;
    __bf16* Wkb = Wqb + wel;
    __bf16* Wvb = Wkb + wel;
    __bf16* Wpb = Wvb + wel;                           // total ~50 MB

    dim3 blk(256);
    // one-shot bf16 conversions
    cvt_bf16_kernel<<<dim3((unsigned)(act / 2048)), blk, 0, stream>>>(x,  xb,  (int)act);
    cvt_bf16_kernel<<<dim3((unsigned)(wel / 2048)), blk, 0, stream>>>(Wq, Wqb, (int)wel);
    cvt_bf16_kernel<<<dim3((unsigned)(wel / 2048)), blk, 0, stream>>>(Wk, Wkb, (int)wel);
    cvt_bf16_kernel<<<dim3((unsigned)(wel / 2048)), blk, 0, stream>>>(Wv, Wvb, (int)wel);
    cvt_bf16_kernel<<<dim3((unsigned)(wel / 2048)), blk, 0, stream>>>(Wp, Wpb, (int)wel);

    dim3 gproj(C_ / 64, (B_ * T_) / 128);              // 16 x 32
    proj_qkv_kernel<<<gproj, blk, 0, stream>>>(xb, Wqb, bq, Qb);
    proj_qkv_kernel<<<gproj, blk, 0, stream>>>(xb, Wkb, bk, Kb);
    proj_qkv_kernel<<<gproj, blk, 0, stream>>>(xb, Wvb, bv, Vb);

    dim3 gattn(T_ / 128, H_, B_);                      // 16 x 16 x 2
    attn_kernel<<<gattn, blk, 0, stream>>>(Qb, Kb, Vb, rel, mask, Yb);

    proj_out_kernel<<<gproj, blk, 0, stream>>>(Yb, Wpb, bp, out);
}
